// CausalAttNet_38534446580225
// MI455X (gfx1250) — compile-verified, matrix operand order checked by reading, not run on hardware
//
#include <hip/hip_runtime.h>
#include <hip/hip_bf16.h>

typedef __bf16 bf16;
typedef __attribute__((ext_vector_type(16))) __bf16 v16bf;
typedef __attribute__((ext_vector_type(8)))  float  v8f;
typedef unsigned long long ull;
typedef __attribute__((ext_vector_type(2))) unsigned long long ullx2;

#define NUM_NODES 32768
#define NUM_EDGES 262144
#define GROUPS    128
#define EPG       2048
#define TOPK      1024

// ---------------------------------------------------------------------------
// Async global->LDS staging (CDNA5 GLOBAL_LOAD_ASYNC_TO_LDS_B128, ASYNCcnt).
// Guarded by __has_builtin; falls back to 16B generic copies.
// Builtin signature (from hipcc diagnostic): first param is
//   int __attribute__((vector_size(16))) __device__ *   (non-const, AS1)
// ---------------------------------------------------------------------------
#ifndef __has_builtin
#define __has_builtin(x) 0
#endif
#if __has_builtin(__builtin_amdgcn_global_load_async_to_lds_b128) && \
    __has_builtin(__builtin_amdgcn_s_wait_asynccnt)
#define USE_ASYNC_CP 1
#else
#define USE_ASYNC_CP 0
#endif

typedef int v4i_ __attribute__((vector_size(16)));
#if USE_ASYNC_CP
typedef __attribute__((address_space(1))) v4i_ gv4i;
typedef __attribute__((address_space(3))) v4i_ lv4i;
#endif

__device__ inline void cp16(void* lds, const void* g) {
#if USE_ASYNC_CP
  __builtin_amdgcn_global_load_async_to_lds_b128(
      (gv4i*)(unsigned long long)g,
      (lv4i*)(unsigned int)(unsigned long long)lds, 0, 0);
#else
  *(ullx2*)lds = *(const ullx2*)g;
#endif
}

__device__ inline void cp16_fence() {
#if USE_ASYNC_CP
  __builtin_amdgcn_s_wait_asynccnt(0);
#endif
}

// ---------------------------------------------------------------------------
// WMMA helpers (CDNA5 wave32, V_WMMA_F32_16X16X32_BF16)
// Per-lane fragment layout (ISA 7.12.2, 16-bit A 16x32 / B 32x16):
//   row = lane&15 ; elements i=0..3 -> K=kb+0..3, 4..7 -> kb+4..7,
//   8..11 -> kb+16..19, 12..15 -> kb+20..23, kb = k0 + 8*(lane>=16)
// ---------------------------------------------------------------------------
__device__ inline v16bf frag_ld(const bf16* base, int row0, int stride, int k0) {
  const int lane = threadIdx.x & 31;
  const bf16* p = base + (size_t)(row0 + (lane & 15)) * stride + k0 + ((lane >> 4) << 3);
  union { v16bf v; ull u[4]; } f;
  f.u[0] = *(const ull*)(p + 0);
  f.u[1] = *(const ull*)(p + 4);
  f.u[2] = *(const ull*)(p + 16);
  f.u[3] = *(const ull*)(p + 20);
  return f.v;
}

__device__ inline v8f wmma_bf16(v16bf a, v16bf b, v8f c) {
  return __builtin_amdgcn_wmma_f32_16x16x32_bf16(false, a, false, b, (short)0, c,
                                                 false, false);
}

// ---------------------------------------------------------------------------
// Format conversion kernels
// ---------------------------------------------------------------------------
__global__ void cvt_x_kernel(const float* __restrict__ x, bf16* __restrict__ xbf, int n) {
  int i = blockIdx.x * 256 + threadIdx.x;
  if (i < n) xbf[i] = (bf16)x[i];
}

__global__ void cvt_w1_kernel(const float* __restrict__ W1, bf16* __restrict__ W1t) {
  int i = blockIdx.x * 256 + threadIdx.x;      // 32768 = 128x256
  int k = i >> 8, n = i & 255;
  W1t[n * 128 + k] = (bf16)W1[i];              // W1t[n][k] = W1[k][n]
}

__global__ void cvt_w2_kernel(const float* __restrict__ W2, bf16* __restrict__ W2t) {
  int i = blockIdx.x * 256 + threadIdx.x;      // 16384 = 256x64
  int k = i >> 6, n = i & 63;
  W2t[n * 256 + k] = (bf16)W2[i];              // W2t[n][k] = W2[k][n]
}

__global__ void zero_kernel(float* __restrict__ p, int n) {
  int i = blockIdx.x * 256 + threadIdx.x;
  if (i < n) p[i] = 0.f;
}

// ---------------------------------------------------------------------------
// Layer 1: h1_pre[e][0:256] = concat(x[src],x[dst]) @ W1 + b1  (bf16 out)
// plus per-column sum / sumsq for batchnorm.  128 edges per block, 8 waves.
// ---------------------------------------------------------------------------
__global__ __launch_bounds__(256) void l1_kernel(
    const int* __restrict__ ei, const bf16* __restrict__ xbf,
    const bf16* __restrict__ W1t, const float* __restrict__ b1,
    bf16* __restrict__ h1, float* __restrict__ gsum, float* __restrict__ gsumsq) {
  __shared__ __align__(16) bf16 At[128][128];   // 32 KB: gathered f12 tile
  __shared__ __align__(16) bf16 Bt[64][128];    // 16 KB: W1^T column chunk
  __shared__ float csum[256];
  __shared__ float csumsq[256];
  const int tid = threadIdx.x;
  const int e0  = blockIdx.x * 128;
  csum[tid] = 0.f; csumsq[tid] = 0.f;
  { // gather endpoint feature rows (async -> LDS): edge t>>1, half t&1
    int el = tid >> 1, half = tid & 1;
    int node = ei[half * NUM_EDGES + e0 + el];
    const bf16* s = xbf + (size_t)node * 64;
    bf16* d = &At[el][half * 64];
#pragma unroll
    for (int i = 0; i < 4; ++i) cp16(d + i * 8, s + i * 8);
  }
  const int w = tid >> 5, lane = tid & 31;
  v16bf a0, a1, a2, a3;
  for (int nc = 0; nc < 4; ++nc) {            // 4 column chunks of 64
    __syncthreads();
    { // async load B chunk: 64 rows (n) x 128 (k)
      int nl = tid >> 2, kq = (tid & 3) * 32;
      const bf16* s = W1t + (size_t)(nc * 64 + nl) * 128 + kq;
      bf16* d = &Bt[nl][kq];
#pragma unroll
      for (int i = 0; i < 4; ++i) cp16(d + i * 8, s + i * 8);
      if (nc < 3)   // global_prefetch_b8 for next W chunk
        __builtin_prefetch(W1t + (size_t)((nc + 1) * 64 + nl) * 128 + kq, 0, 1);
    }
    cp16_fence();   // s_wait_asynccnt 0 (covers A tile on first iteration)
    __syncthreads();
    if (nc == 0) {  // A fragments persist across all column chunks
      a0 = frag_ld(&At[0][0], w * 16, 128, 0);
      a1 = frag_ld(&At[0][0], w * 16, 128, 32);
      a2 = frag_ld(&At[0][0], w * 16, 128, 64);
      a3 = frag_ld(&At[0][0], w * 16, 128, 96);
    }
#pragma unroll
    for (int nt = 0; nt < 4; ++nt) {
      v8f acc = {0.f,0.f,0.f,0.f,0.f,0.f,0.f,0.f};
      v16bf bb0 = frag_ld(&Bt[0][0], nt * 16, 128, 0);
      v16bf bb1 = frag_ld(&Bt[0][0], nt * 16, 128, 32);
      v16bf bb2 = frag_ld(&Bt[0][0], nt * 16, 128, 64);
      v16bf bb3 = frag_ld(&Bt[0][0], nt * 16, 128, 96);
      acc = wmma_bf16(a0, bb0, acc);
      acc = wmma_bf16(a1, bb1, acc);
      acc = wmma_bf16(a2, bb2, acc);
      acc = wmma_bf16(a3, bb3, acc);
      const int col = nc * 64 + nt * 16 + (lane & 15);
      const float bias = b1[col];
      // one per-lane base pointer; r*256 folds into the store ioffset
      bf16* hp = h1 + ((size_t)(e0 + w * 16 + ((lane >> 4) << 3)) * 256 + col);
      float s = 0.f, s2 = 0.f;
#pragma unroll
      for (int r = 0; r < 8; ++r) {   // C layout: M = r + 8*(lane>=16), N = lane&15
        float v = acc[r] + bias;
        hp[(size_t)r * 256] = (bf16)v;
        s += v; s2 += v * v;
      }
      atomicAdd(&csum[col], s);
      atomicAdd(&csumsq[col], s2);
    }
  }
  __syncthreads();
  atomicAdd(&gsum[tid], csum[tid]);
  atomicAdd(&gsumsq[tid], csumsq[tid]);
}

// Fold batchnorm params: scale = g*rsqrt(var+eps), shift = be - mu*scale
__global__ void bn_fin_kernel(const float* __restrict__ sum, const float* __restrict__ sumsq,
                              const float* __restrict__ g, const float* __restrict__ be,
                              float* __restrict__ scale, float* __restrict__ shift,
                              int C, float inv_n) {
  int c = blockIdx.x * 64 + threadIdx.x;
  if (c < C) {
    float mu  = sum[c] * inv_n;
    float var = sumsq[c] * inv_n - mu * mu;
    float sc  = g[c] * rsqrtf(var + 1e-5f);
    scale[c] = sc;
    shift[c] = be[c] - mu * sc;
  }
}

// ---------------------------------------------------------------------------
// Layer 2: h2_pre[e][0:64] = relu(bn1(h1_pre)) @ W2 + b2  (f32 out) + stats
// ---------------------------------------------------------------------------
__global__ __launch_bounds__(256) void l2_kernel(
    const bf16* __restrict__ h1, const bf16* __restrict__ W2t,
    const float* __restrict__ sc1, const float* __restrict__ sh1,
    const float* __restrict__ b2, float* __restrict__ h2,
    float* __restrict__ gsum, float* __restrict__ gsumsq) {
  __shared__ __align__(16) bf16 Bt[64][256];   // 32 KB: full W2^T
  __shared__ __align__(16) bf16 As[128][64];   // 16 KB: staged bn+relu A chunk
  __shared__ float sc[256], sh[256];
  __shared__ float csum[64], csumsq[64];
  const int tid = threadIdx.x;
  const int e0  = blockIdx.x * 128;
  sc[tid] = sc1[tid]; sh[tid] = sh1[tid];
  if (tid < 64) { csum[tid] = 0.f; csumsq[tid] = 0.f; }
  { // async load whole W2^T (64 x 256)
    int nl = tid >> 2, kq = (tid & 3) * 64;
    const bf16* s = W2t + (size_t)nl * 256 + kq;
    bf16* d = &Bt[nl][kq];
#pragma unroll
    for (int i = 0; i < 8; ++i) cp16(d + i * 8, s + i * 8);
  }
  cp16_fence();
  const int w = tid >> 5, lane = tid & 31;
  v8f acc[4];
  const v8f vz = {0.f,0.f,0.f,0.f,0.f,0.f,0.f,0.f};
#pragma unroll
  for (int nt = 0; nt < 4; ++nt) acc[nt] = vz;
  for (int kc = 0; kc < 4; ++kc) {             // K in chunks of 64
    __syncthreads();
    { // stage A chunk with fused bn1 + relu: thread -> edge t>>1, 32 cols
      int el = tid >> 1, part = (tid & 1) * 32;
      const ull* s = (const ull*)(h1 + (size_t)(e0 + el) * 256 + kc * 64 + part);
      ull* d = (ull*)&As[el][part];
#pragma unroll
      for (int i = 0; i < 8; ++i) {
        union { ull u; bf16 b[4]; } v;
        v.u = s[i];
#pragma unroll
        for (int q = 0; q < 4; ++q) {
          int col = kc * 64 + part + i * 4 + q;
          float f = (float)v.b[q] * sc[col] + sh[col];
          v.b[q] = (bf16)(f > 0.f ? f : 0.f);
        }
        d[i] = v.u;
      }
    }
    __syncthreads();
#pragma unroll
    for (int kk = 0; kk < 64; kk += 32) {
      v16bf a = frag_ld(&As[0][0], w * 16, 64, kk);
#pragma unroll
      for (int nt = 0; nt < 4; ++nt) {
        v16bf b = frag_ld(&Bt[0][0], nt * 16, 256, kc * 64 + kk);
        acc[nt] = wmma_bf16(a, b, acc[nt]);
      }
    }
  }
#pragma unroll
  for (int nt = 0; nt < 4; ++nt) {
    int col = nt * 16 + (lane & 15);
    float bias = b2[col];
    float* hp = h2 + ((size_t)(e0 + w * 16 + ((lane >> 4) << 3)) * 64 + col);
    float s = 0.f, s2 = 0.f;
#pragma unroll
    for (int r = 0; r < 8; ++r) {
      float v = acc[nt][r] + bias;
      hp[(size_t)r * 64] = v;
      s += v; s2 += v * v;
    }
    atomicAdd(&csum[col], s);
    atomicAdd(&csumsq[col], s2);
  }
  __syncthreads();
  if (tid < 64) {
    atomicAdd(&gsum[tid], csum[tid]);
    atomicAdd(&gsumsq[tid], csumsq[tid]);
  }
}

// Layer 3: score[e] = relu(bn2(h2_pre[e])) . W3 + b3
__global__ __launch_bounds__(256) void score_kernel(
    const float* __restrict__ h2, const float* __restrict__ sc2,
    const float* __restrict__ sh2, const float* __restrict__ W3,
    const float* __restrict__ b3, float* __restrict__ score) {
  __shared__ float sc[64], sh[64], w3[64];
  int tid = threadIdx.x;
  if (tid < 64) { sc[tid] = sc2[tid]; sh[tid] = sh2[tid]; w3[tid] = W3[tid]; }
  __syncthreads();
  int e = blockIdx.x * 256 + tid;
  const float* row = h2 + (size_t)e * 64;
  float s = b3[0];
#pragma unroll
  for (int c = 0; c < 64; ++c) {
    float v = row[c] * sc[c] + sh[c];
    s += (v > 0.f ? v : 0.f) * w3[c];
  }
  score[e] = s;
}

// ---------------------------------------------------------------------------
// Per-group bitonic sort (desc score, stable by index) -> top-K split.
// One block per group, 2048 keys in LDS.
// ---------------------------------------------------------------------------
__global__ __launch_bounds__(1024) void sort_kernel(
    const float* __restrict__ score, int* __restrict__ causal_idx,
    int* __restrict__ conf_idx) {
  __shared__ ull keys[EPG];
  const int g = blockIdx.x, tid = threadIdx.x;
  for (int i = tid; i < EPG; i += 1024) {
    unsigned u = __float_as_uint(score[g * EPG + i]);
    u = (u & 0x80000000u) ? ~u : (u | 0x80000000u);  // monotone ascending map
    u = ~u;                                          // descending
    keys[i] = ((ull)u << 32) | (unsigned)i;          // low bits: stable index
  }
  for (int k = 2; k <= EPG; k <<= 1)
    for (int j = k >> 1; j > 0; j >>= 1) {
      __syncthreads();
      for (int i = tid; i < EPG; i += 1024) {
        int ixj = i ^ j;
        if (ixj > i) {
          bool up = ((i & k) == 0);
          ull a = keys[i], b = keys[ixj];
          if ((a > b) == up) { keys[i] = b; keys[ixj] = a; }
        }
      }
    }
  __syncthreads();
  for (int p = tid; p < EPG; p += 1024) {
    int e = g * EPG + (int)(keys[p] & 0xFFFFFFFFull);
    if (p < TOPK) causal_idx[g * TOPK + p] = e;
    else          conf_idx[g * TOPK + (p - TOPK)] = e;
  }
}

// Mark nodes present in the selected edge set
__global__ void mark_kernel(const int* __restrict__ idx, const int* __restrict__ ei,
                            int* __restrict__ mark) {
  int i = blockIdx.x * 256 + threadIdx.x;            // < 131072
  int e = idx[i];
  mark[ei[e]] = 1;
  mark[ei[NUM_EDGES + e]] = 1;
}

// Exclusive prefix scan of 32768 flags in one workgroup + scatter sub[]
__global__ __launch_bounds__(1024) void scan_kernel(const int* __restrict__ mark,
                                                    int* __restrict__ psum,
                                                    int* __restrict__ sub) {
  __shared__ int wsums[1024];
  __shared__ int gsums[33];
  const int tid = threadIdx.x;
  const int base = tid * 32;
  int loc[32]; int s = 0;
#pragma unroll
  for (int i = 0; i < 32; ++i) { loc[i] = mark[base + i]; s += loc[i]; }
  wsums[tid] = s;
  __syncthreads();
  if (tid < 32) {
    int t = 0;
    for (int q = 0; q < 32; ++q) t += wsums[tid * 32 + q];
    gsums[tid] = t;
  }
  __syncthreads();
  if (tid == 0) {
    int acc = 0;
    for (int q = 0; q < 32; ++q) { int t = gsums[q]; gsums[q] = acc; acc += t; }
    gsums[32] = acc;
  }
  __syncthreads();
  int run = gsums[tid >> 5];
  for (int q = (tid & ~31); q < tid; ++q) run += wsums[q];
#pragma unroll
  for (int i = 0; i < 32; ++i) {
    psum[base + i] = run;
    if (loc[i]) sub[run] = base + i;   // sub pre-zeroed -> padding stays node 0
    run += loc[i];
  }
}

// out_x[p] = x[sub[p]], out_batch[p] = batch[sub[p]]  (4 nodes per block)
__global__ void gather_x_kernel(const int* __restrict__ sub, const float* __restrict__ x,
                                const int* __restrict__ batch,
                                float* __restrict__ out_x, float* __restrict__ out_b) {
  int p = blockIdx.x * 4 + (threadIdx.x >> 6);
  int c = threadIdx.x & 63;
  int v = sub[p];
  out_x[(size_t)p * 64 + c] = x[(size_t)v * 64 + c];
  if (c == 0) out_b[p] = (float)batch[v];
}

// Relabeled edge indices + weights
__global__ void edge_out_kernel(const int* __restrict__ idx, const int* __restrict__ ei,
                                const int* __restrict__ psum, const float* __restrict__ score,
                                float* __restrict__ out_ei, float* __restrict__ out_w,
                                float sign) {
  int i = blockIdx.x * 256 + threadIdx.x;            // < 131072
  int e = idx[i];
  out_ei[i]          = (float)psum[ei[e]];
  out_ei[131072 + i] = (float)psum[ei[NUM_EDGES + e]];
  out_w[i] = sign * score[e];
}

// ---------------------------------------------------------------------------
extern "C" void kernel_launch(void* const* d_in, const int* in_sizes, int n_in,
                              void* d_out, int out_size, void* d_ws, size_t ws_size,
                              hipStream_t stream) {
  const float* x     = (const float*)d_in[0];
  const int*   ei    = (const int*)d_in[1];
  const int*   batch = (const int*)d_in[2];
  const float* W1    = (const float*)d_in[3];
  const float* b1    = (const float*)d_in[4];
  const float* g1    = (const float*)d_in[5];
  const float* be1   = (const float*)d_in[6];
  const float* W2    = (const float*)d_in[7];
  const float* b2    = (const float*)d_in[8];
  const float* g2    = (const float*)d_in[9];
  const float* be2   = (const float*)d_in[10];
  const float* W3    = (const float*)d_in[11];
  const float* b3    = (const float*)d_in[12];
  float* out = (float*)d_out;
  (void)in_sizes; (void)n_in; (void)out_size; (void)ws_size;

  char* ws = (char*)d_ws;
  size_t off = 0;
  auto alloc = [&](size_t bytes) -> void* {
    off = (off + 255) & ~(size_t)255;
    void* p = ws + off; off += bytes; return p;
  };
  bf16*  xbf   = (bf16*) alloc((size_t)NUM_NODES * 64 * 2);
  bf16*  W1t   = (bf16*) alloc(256 * 128 * 2);
  bf16*  W2t   = (bf16*) alloc(64 * 256 * 2);
  bf16*  h1    = (bf16*) alloc((size_t)NUM_EDGES * 256 * 2);   // 128 MB
  float* h2    = (float*)alloc((size_t)NUM_EDGES * 64 * 4);    //  64 MB
  float* stats = (float*)alloc(1280 * 4);
  float *sum1 = stats,        *sumsq1 = stats + 256,
        *scale1 = stats + 512, *shift1 = stats + 768;
  float *sum2 = stats + 1024, *sumsq2 = stats + 1088,
        *scale2 = stats + 1152, *shift2 = stats + 1216;
  int* cidx  = (int*)alloc((size_t)131072 * 4);
  int* fidx  = (int*)alloc((size_t)131072 * 4);
  int* cmark = (int*)alloc(NUM_NODES * 4);
  int* cpsum = (int*)alloc(NUM_NODES * 4);
  int* csub  = (int*)alloc(NUM_NODES * 4);
  int* fmark = (int*)alloc(NUM_NODES * 4);
  int* fpsum = (int*)alloc(NUM_NODES * 4);
  int* fsub  = (int*)alloc(NUM_NODES * 4);

  // d_out layout (floats), reference return order with None skipped:
  float* o_cx  = out;               // causal_x     2,097,152
  float* o_cei = out + 2097152;     // causal_ei2     262,144
  float* o_cw  = out + 2359296;     // causal_w       131,072
  float* o_cb  = out + 2490368;     // causal_batch    32,768
  float* o_fx  = out + 2523136;     // conf_x       2,097,152
  float* o_fei = out + 4620288;     // conf_ei2       262,144
  float* o_fw  = out + 4882432;     // conf_w         131,072
  float* o_fb  = out + 5013504;     // conf_batch      32,768
  float* o_sc  = out + 5046272;     // edge_score     262,144

  zero_kernel<<<5,   256, 0, stream>>>(stats, 1280);
  zero_kernel<<<128, 256, 0, stream>>>((float*)cmark, NUM_NODES);
  zero_kernel<<<128, 256, 0, stream>>>((float*)csub,  NUM_NODES);
  zero_kernel<<<128, 256, 0, stream>>>((float*)fmark, NUM_NODES);
  zero_kernel<<<128, 256, 0, stream>>>((float*)fsub,  NUM_NODES);

  cvt_x_kernel <<<8192, 256, 0, stream>>>(x, xbf, NUM_NODES * 64);
  cvt_w1_kernel<<<128,  256, 0, stream>>>(W1, W1t);
  cvt_w2_kernel<<<64,   256, 0, stream>>>(W2, W2t);

  l1_kernel<<<2048, 256, 0, stream>>>(ei, xbf, W1t, b1, h1, sum1, sumsq1);
  bn_fin_kernel<<<4, 64, 0, stream>>>(sum1, sumsq1, g1, be1, scale1, shift1,
                                      256, 1.0f / (float)NUM_EDGES);
  l2_kernel<<<2048, 256, 0, stream>>>(h1, W2t, scale1, shift1, b2, h2, sum2, sumsq2);
  bn_fin_kernel<<<1, 64, 0, stream>>>(sum2, sumsq2, g2, be2, scale2, shift2,
                                      64, 1.0f / (float)NUM_EDGES);
  score_kernel<<<1024, 256, 0, stream>>>(h2, scale2, shift2, W3, b3, o_sc);

  sort_kernel<<<GROUPS, 1024, 0, stream>>>(o_sc, cidx, fidx);
  mark_kernel<<<512, 256, 0, stream>>>(cidx, ei, cmark);
  mark_kernel<<<512, 256, 0, stream>>>(fidx, ei, fmark);
  scan_kernel<<<1, 1024, 0, stream>>>(cmark, cpsum, csub);
  scan_kernel<<<1, 1024, 0, stream>>>(fmark, fpsum, fsub);
  gather_x_kernel<<<8192, 256, 0, stream>>>(csub, x, batch, o_cx, o_cb);
  gather_x_kernel<<<8192, 256, 0, stream>>>(fsub, x, batch, o_fx, o_fb);
  edge_out_kernel<<<512, 256, 0, stream>>>(cidx, ei, cpsum, o_sc, o_cei, o_cw,  1.0f);
  edge_out_kernel<<<512, 256, 0, stream>>>(fidx, ei, fpsum, o_sc, o_fei, o_fw, -1.0f);
}
